// GLOBE_35150012350798
// MI455X (gfx1250) — matrix operand order repaired and checked by reading, not compiled, not run on hardware
//
#include <hip/hip_runtime.h>
#include <hip/hip_bf16.h>

// ---------------- problem constants ----------------
#define NB   2
#define NS   2
#define NF   512
#define NPTS 1024
#define NLS  12
#define NLV  6
#define HH   64
#define FF   29
#define NTGT 1024            // NB*NF == NPTS

typedef __attribute__((ext_vector_type(16))) _Float16 v16h;
typedef __attribute__((ext_vector_type(8)))  float    v8f;

// ---------------- LDS byte map ----------------
#define OFF_W1   0u          // [64n][32k] f16  (4096 B, K padded 29->32)
#define OFF_W2   4096u       // [2kc][64n][32k] (8192 B)
#define OFF_W3   12288u      // [2kc][64n][32k] (8192 B)
#define OFF_WO   20480u      // [2kc][32n][32k] (4096 B, N padded ->32)
#define OFF_B1   24576u      // 64 f32
#define OFF_B2   24832u
#define OFF_B3   25088u
#define OFF_BO   25344u      // 32 f32
#define OFF_WAVE 25472u
// per-wave region: 32 rows now (two 16-row A tiles)
#define FEATS_OFF 0u         // 32 rows x 32 f16 = 2048 B
#define ACT1_OFF  2048u      // 32 rows x 64 f16 = 4096 B
#define ACT2_OFF  6144u      // 4096 B
#define OUTB_OFF  10240u     // 32 rows x 32 f32 = 4096 B
#define WAVE_BYTES 14336u
#define SMEM_BYTES (25472u + 2u*14336u)   // 54144 B

#define BLOCK 64             // 2 waves x 32 lanes

// ---------------- fragment helpers ----------------
union FragU { v16h v; uint4 q[2]; };
union FeatPack { _Float16 h[32]; uint4 q[4]; };

// A fragment (16x32 f16, row-major rows in LDS). Lane<16: K0..7 then K16..23;
// lane>=16: K8..15 then K24..31  (ISA 7.12.2 16-bit A layout).
__device__ __forceinline__ v16h ldsA(const unsigned char* p, int row, int rowStrideB,
                                     int kcByte, int lane) {
  const int half = (lane >> 4) & 1;
  const unsigned off = (unsigned)(row * rowStrideB + kcByte + half * 16);
  FragU u;
  u.q[0] = *(const uint4*)(p + off);
  u.q[1] = *(const uint4*)(p + off + 32);
  return u.v;
}

// B fragment (32x16 f16, LDS layout [n][k] with 64B per column).
// Lane<16: K0..7 then K16..23 of its column; lane>=16: K8..15 then K24..31.
__device__ __forceinline__ v16h ldsB(const unsigned char* p, unsigned colBase, int lane) {
  const int half = (lane >> 4) & 1;
  const unsigned off = colBase + (unsigned)(half * 16);
  FragU u;
  u.q[0] = *(const uint4*)(p + off);
  u.q[1] = *(const uint4*)(p + off + 32);
  return u.v;
}

__device__ __forceinline__ v8f wmma_f16(v16h a, v16h b, v8f c) {
  return __builtin_amdgcn_wmma_f32_16x16x32_f16(false, a, false, b, (short)0, c, false, false);
}

// fast silu: one v_exp_f32 + one v_rcp_f32
__device__ __forceinline__ float siluf(float x) {
  return x * __builtin_amdgcn_rcpf(1.0f + __expf(-x));
}

// one hidden layer, dual 16-row tiles (rows 0-15 and 16-31):
// B fragments loaded once, fed to two independent WMMAs (fills hazard slots).
__device__ __forceinline__ void mlp_hidden2(unsigned char* smem, unsigned char* wbase,
                                            unsigned inOff, int inStrideB, int nKc,
                                            unsigned wOff, unsigned wKcStride,
                                            const float* biasL, unsigned outOff, int lane) {
  const int c = lane & 15;
  const int half = (lane >> 4) & 1;
  const int r0 = lane & 15;
  const v8f zero = {};
  v8f acc0[4], acc1[4];
  {
    const v16h a0 = ldsA(wbase + inOff, r0, inStrideB, 0, lane);
    const v16h a1 = ldsA(wbase + inOff, r0 + 16, inStrideB, 0, lane);
#pragma unroll
    for (int nt = 0; nt < 4; ++nt) {
      const v16h bf = ldsB(smem, wOff + (unsigned)(nt * 16 + c) * 64, lane);
      acc0[nt] = wmma_f16(a0, bf, zero);
      acc1[nt] = wmma_f16(a1, bf, zero);
    }
  }
  for (int kc = 1; kc < nKc; ++kc) {
    const v16h a0 = ldsA(wbase + inOff, r0, inStrideB, kc * 64, lane);
    const v16h a1 = ldsA(wbase + inOff, r0 + 16, inStrideB, kc * 64, lane);
#pragma unroll
    for (int nt = 0; nt < 4; ++nt) {
      const v16h bf = ldsB(smem, wOff + (unsigned)kc * wKcStride + (unsigned)(nt * 16 + c) * 64, lane);
      acc0[nt] = wmma_f16(a0, bf, acc0[nt]);
      acc1[nt] = wmma_f16(a1, bf, acc1[nt]);
    }
  }
  _Float16* act = (_Float16*)(wbase + outOff);
#pragma unroll
  for (int nt = 0; nt < 4; ++nt) {
    const float bv = biasL[nt * 16 + c];
#pragma unroll
    for (int r = 0; r < 8; ++r) {
      act[(half * 8 + r) * 64 + nt * 16 + c] = (_Float16)siluf(acc0[nt][r] + bv);
      act[(16 + half * 8 + r) * 64 + nt * 16 + c] = (_Float16)siluf(acc1[nt][r] + bv);
    }
  }
}

// ---------------- main pass kernel ----------------
template <bool FINAL>
__global__ void __launch_bounds__(BLOCK)
globe_pass(const float* __restrict__ targets,      // 1024 x 3
           const float* __restrict__ centroids,    // NB*NF x 3
           const float* __restrict__ nrmW,         // NB*NF x 3 (normalized)
           const float* __restrict__ areas,        // NB*NF
           const float* __restrict__ refL,         // NS
           const float* __restrict__ scalIn,       // NTGT x 12
           const float* __restrict__ vecsIn,       // NTGT x 18
           const float* __restrict__ vnIn,         // NTGT x 6
           const float* __restrict__ gW1, const float* __restrict__ gB1,
           const float* __restrict__ gW2, const float* __restrict__ gB2,
           const float* __restrict__ gW3, const float* __restrict__ gB3,
           const float* __restrict__ gWo, const float* __restrict__ gBo,
           float* __restrict__ outA, float* __restrict__ outB) {
  constexpr int OUTC = FINAL ? 3 : 24;
  constexpr int NT_OUT = FINAL ? 1 : 2;

  __shared__ __align__(16) unsigned char smem[SMEM_BYTES];

  const int tid = threadIdx.x;
  const int lane = tid & 31;
  const int wave = tid >> 5;
  const int bs = blockIdx.x & 3;
  const int tt = blockIdx.x >> 2;
  const int b = bs >> 1, s = bs & 1;
  const int wofs = b * NS + s;

  gW1 += wofs * FF * HH;   gB1 += wofs * HH;
  gW2 += wofs * HH * HH;   gB2 += wofs * HH;
  gW3 += wofs * HH * HH;   gB3 += wofs * HH;
  gWo += wofs * HH * OUTC; gBo += wofs * OUTC;

  _Float16* w1 = (_Float16*)(smem + OFF_W1);
  _Float16* w2 = (_Float16*)(smem + OFF_W2);
  _Float16* w3 = (_Float16*)(smem + OFF_W3);
  _Float16* wo = (_Float16*)(smem + OFF_WO);
  float* b1L = (float*)(smem + OFF_B1);
  float* b2L = (float*)(smem + OFF_B2);
  float* b3L = (float*)(smem + OFF_B3);
  float* boL = (float*)(smem + OFF_BO);

  // ---- stage weights: zero pads, then transposed fill into [n][k] layout ----
  for (int i = tid; i < 2048; i += BLOCK) w1[i] = (_Float16)0.f;
  for (int i = tid; i < 2048; i += BLOCK) wo[i] = (_Float16)0.f;
  for (int i = tid; i < 32; i += BLOCK) boL[i] = 0.f;
  __syncthreads();
  for (int i = tid; i < FF * HH; i += BLOCK) {
    const int k = i >> 6, n = i & 63;
    w1[n * 32 + k] = (_Float16)gW1[i];
  }
  for (int i = tid; i < HH * HH; i += BLOCK) {
    const int k = i >> 6, n = i & 63;
    w2[(k >> 5) * 2048 + n * 32 + (k & 31)] = (_Float16)gW2[i];
  }
  for (int i = tid; i < HH * HH; i += BLOCK) {
    const int k = i >> 6, n = i & 63;
    w3[(k >> 5) * 2048 + n * 32 + (k & 31)] = (_Float16)gW3[i];
  }
  for (int i = tid; i < HH * OUTC; i += BLOCK) {
    const int k = i / OUTC, n = i % OUTC;
    wo[(k >> 5) * 1024 + n * 32 + (k & 31)] = (_Float16)gWo[i];
  }
  for (int i = tid; i < HH; i += BLOCK) { b1L[i] = gB1[i]; b2L[i] = gB2[i]; b3L[i] = gB3[i]; }
  for (int i = tid; i < OUTC; i += BLOCK) boL[i] = gBo[i];
  __syncthreads();

  unsigned char* wbase = smem + OFF_WAVE + (unsigned)wave * WAVE_BYTES;
  const float invL = __builtin_amdgcn_rcpf(refL[s]);
  const int t = tt * 64 + wave * 32 + lane;   // every lane owns one target row

  const float tx = targets[t * 3 + 0];
  const float ty = targets[t * 3 + 1];
  const float tz = targets[t * 3 + 2];

  float nsAcc[12] = {0};
  float nvAcc[18] = {0};
  float pAcc = 0.f;
  float vAcc[3] = {0.f, 0.f, 0.f};

#pragma unroll 1
  for (int ss = 0; ss < NF; ++ss) {
    const int src = b * NF + ss;
    // prefetch next source's per-source rows (global_prefetch_b8)
    __builtin_prefetch(&centroids[(src + 1) * 3], 0, 3);
    __builtin_prefetch(&scalIn[(src + 1) * NLS], 0, 3);

    const float px = centroids[src * 3 + 0], py = centroids[src * 3 + 1], pz = centroids[src * 3 + 2];
    const float snx = nrmW[src * 3 + 0], sny = nrmW[src * 3 + 1], snz = nrmW[src * 3 + 2];
    const float w = areas[src];

    // ---- per-pair feature generation: one row per lane (fully convergent) ----
    const float rx = tx - px, ry = ty - py, rz = tz - pz;
    const float r2 = rx * rx + ry * ry + rz * rz + 1e-16f;
    const float ri = __builtin_amdgcn_rsqf(r2);   // 1/sqrt(r2)
    const float r = r2 * ri;                      // sqrt(r2)
    const float rhx = rx * ri, rhy = ry * ri, rhz = rz * ri;
    const float cv = rhx * snx + rhy * sny + rhz * snz;
    {
      FeatPack fp;
      fp.h[0] = (_Float16)__logf(r * invL);
      fp.h[1] = (_Float16)1.0f;
      fp.h[2] = (_Float16)cv;
      fp.h[3] = (_Float16)(0.5f * (3.0f * cv * cv - 1.0f));
      fp.h[4] = (_Float16)(0.5f * (5.0f * cv * cv * cv - 3.0f * cv));
#pragma unroll
      for (int k = 0; k < NLS; ++k) fp.h[5 + k] = (_Float16)scalIn[src * NLS + k];
#pragma unroll
      for (int v = 0; v < NLV; ++v) {
        const float vx = vecsIn[src * 18 + v * 3 + 0];
        const float vy = vecsIn[src * 18 + v * 3 + 1];
        const float vz = vecsIn[src * 18 + v * 3 + 2];
        fp.h[17 + v] = (_Float16)(rhx * vx + rhy * vy + rhz * vz);
        fp.h[23 + v] = (_Float16)vnIn[src * NLV + v];
      }
      fp.h[29] = fp.h[30] = fp.h[31] = (_Float16)0.f;
      uint4* frow = (uint4*)(wbase + FEATS_OFF + (unsigned)lane * 64);
      frow[0] = fp.q[0]; frow[1] = fp.q[1]; frow[2] = fp.q[2]; frow[3] = fp.q[3];
    }

    // ---- MLP: 32 rows through 29->64->64->64 (dual-tile WMMA) ----
    mlp_hidden2(smem, wbase, FEATS_OFF, 64, 1, OFF_W1, 0u, b1L, ACT1_OFF, lane);
    mlp_hidden2(smem, wbase, ACT1_OFF, 128, 2, OFF_W2, 4096u, b2L, ACT2_OFF, lane);
    mlp_hidden2(smem, wbase, ACT2_OFF, 128, 2, OFF_W3, 4096u, b3L, ACT1_OFF, lane);

    // ---- output layer (C starts as inline 0; bias added in epilogue) ----
    {
      const int c = lane & 15;
      const int half = (lane >> 4) & 1;
      const int r0 = lane & 15;
      const v8f zero = {};
      v8f oacc0[NT_OUT], oacc1[NT_OUT];
      {
        const v16h a0 = ldsA(wbase + ACT1_OFF, r0, 128, 0, lane);
        const v16h a1 = ldsA(wbase + ACT1_OFF, r0 + 16, 128, 0, lane);
#pragma unroll
        for (int nt = 0; nt < NT_OUT; ++nt) {
          const v16h bf = ldsB(smem, OFF_WO + (unsigned)(nt * 16 + c) * 64, lane);
          oacc0[nt] = wmma_f16(a0, bf, zero);
          oacc1[nt] = wmma_f16(a1, bf, zero);
        }
      }
      {
        const v16h a0 = ldsA(wbase + ACT1_OFF, r0, 128, 64, lane);
        const v16h a1 = ldsA(wbase + ACT1_OFF, r0 + 16, 128, 64, lane);
#pragma unroll
        for (int nt = 0; nt < NT_OUT; ++nt) {
          const v16h bf = ldsB(smem, OFF_WO + 2048u + (unsigned)(nt * 16 + c) * 64, lane);
          oacc0[nt] = wmma_f16(a0, bf, oacc0[nt]);
          oacc1[nt] = wmma_f16(a1, bf, oacc1[nt]);
        }
      }
      float* ob = (float*)(wbase + OUTB_OFF);
#pragma unroll
      for (int nt = 0; nt < NT_OUT; ++nt) {
        const float bv = boL[nt * 16 + c];
#pragma unroll
        for (int r = 0; r < 8; ++r) {
          ob[(half * 8 + r) * 32 + nt * 16 + c] = oacc0[nt][r] + bv;
          ob[(16 + half * 8 + r) * 32 + nt * 16 + c] = oacc1[nt][r] + bv;
        }
      }
    }

    // ---- per-target aggregation: one row per lane, vectorized LDS reads ----
    {
      const uint4* o4 = (const uint4*)(wbase + OUTB_OFF + (unsigned)lane * 128);
      if constexpr (FINAL) {
        union { float f[4]; uint4 q; } ob0;
        ob0.q = o4[0];
        pAcc = fmaf(ob0.f[0], w, pAcc);
        const float a1 = ob0.f[1] * w, a2 = ob0.f[2] * w;
        vAcc[0] += a1 * rhx + a2 * snx;
        vAcc[1] += a1 * rhy + a2 * sny;
        vAcc[2] += a1 * rhz + a2 * snz;
      } else {
        union { float f[24]; uint4 q[6]; } ob;
#pragma unroll
        for (int i = 0; i < 6; ++i) ob.q[i] = o4[i];
#pragma unroll
        for (int k = 0; k < 12; ++k) nsAcc[k] = fmaf(ob.f[k], w, nsAcc[k]);
#pragma unroll
        for (int v = 0; v < 6; ++v) {
          const float a1 = ob.f[12 + v] * w, a2 = ob.f[18 + v] * w;
          nvAcc[v * 3 + 0] += a1 * rhx + a2 * snx;
          nvAcc[v * 3 + 1] += a1 * rhy + a2 * sny;
          nvAcc[v * 3 + 2] += a1 * rhz + a2 * snz;
        }
      }
    }
  }

  // ---- cross-block reduction via f32 global atomics (all lanes) ----
  if constexpr (FINAL) {
    atomicAdd(&outA[t], pAcc);
#pragma unroll
    for (int d = 0; d < 3; ++d) atomicAdd(&outB[t * 3 + d], vAcc[d]);
  } else {
#pragma unroll
    for (int k = 0; k < 12; ++k) atomicAdd(&outA[t * 12 + k], nsAcc[k]);
#pragma unroll
    for (int j = 0; j < 18; ++j) atomicAdd(&outB[t * 18 + j], nvAcc[j]);
  }
}

// ---------------- helper kernels ----------------
#define ZERO_FLOATS 114688   // scal0..vn2 (3*36864) + pacc(1024) + vacc(3072)

__global__ void prep_kernel(const float* __restrict__ normals, float* __restrict__ ws) {
  const int i = blockIdx.x * blockDim.x + threadIdx.x;
  if (i < ZERO_FLOATS) ws[3072 + i] = 0.f;
  if (i < NTGT) {
    const float x = normals[i * 3 + 0], y = normals[i * 3 + 1], z = normals[i * 3 + 2];
    const float inv = __builtin_amdgcn_rsqf(x * x + y * y + z * z);
    ws[i * 3 + 0] = x * inv;
    ws[i * 3 + 1] = y * inv;
    ws[i * 3 + 2] = z * inv;
  }
}

__global__ void vn_kernel(const float* __restrict__ vecs, const float* __restrict__ nrm,
                          float* __restrict__ vn) {
  const int i = blockIdx.x * blockDim.x + threadIdx.x;
  if (i < NTGT * NLV) {
    const int f = i / NLV, v = i % NLV;
    vn[i] = vecs[f * 18 + v * 3 + 0] * nrm[f * 3 + 0] +
            vecs[f * 18 + v * 3 + 1] * nrm[f * 3 + 1] +
            vecs[f * 18 + v * 3 + 2] * nrm[f * 3 + 2];
  }
}

__global__ void calib_kernel(const float* __restrict__ pacc, const float* __restrict__ vacc,
                             const float* __restrict__ calib, float* __restrict__ out) {
  const int i = blockIdx.x * blockDim.x + threadIdx.x;
  if (i < NPTS) {
    const float c0 = calib[0], c1 = calib[1], c2 = calib[2];
    out[i * 4 + 0] = c0 * pacc[i] + c1;
    out[i * 4 + 1] = c2 * vacc[i * 3 + 0];
    out[i * 4 + 2] = c2 * vacc[i * 3 + 1];
    out[i * 4 + 3] = c2 * vacc[i * 3 + 2];
  }
}

// ---------------- host launcher ----------------
extern "C" void kernel_launch(void* const* d_in, const int* in_sizes, int n_in,
                              void* d_out, int out_size, void* d_ws, size_t ws_size,
                              hipStream_t stream) {
  (void)in_sizes; (void)n_in; (void)out_size; (void)ws_size;
  const float* pred      = (const float*)d_in[0];
  const float* centroids = (const float*)d_in[1];
  const float* normals   = (const float*)d_in[2];
  const float* areas     = (const float*)d_in[3];
  const float* refL      = (const float*)d_in[4];
  const float* cW1 = (const float*)d_in[5];
  const float* cb1 = (const float*)d_in[6];
  const float* cW2 = (const float*)d_in[7];
  const float* cb2 = (const float*)d_in[8];
  const float* cW3 = (const float*)d_in[9];
  const float* cb3 = (const float*)d_in[10];
  const float* cWo = (const float*)d_in[11];
  const float* cbo = (const float*)d_in[12];
  const float* fW1 = (const float*)d_in[13];
  const float* fb1 = (const float*)d_in[14];
  const float* fW2 = (const float*)d_in[15];
  const float* fb2 = (const float*)d_in[16];
  const float* fW3 = (const float*)d_in[17];
  const float* fb3 = (const float*)d_in[18];
  const float* fWo = (const float*)d_in[19];
  const float* fbo = (const float*)d_in[20];
  const float* calib = (const float*)d_in[21];

  float* wsf   = (float*)d_ws;
  float* nrmW  = wsf;                    // 3072
  float* scal0 = wsf + 3072;             // 12288
  float* vecs0 = scal0 + 12288;          // 18432
  float* vn0   = vecs0 + 18432;          // 6144
  float* scal1 = vn0 + 6144;
  float* vecs1 = scal1 + 12288;
  float* vn1   = vecs1 + 18432;
  float* scal2 = vn1 + 6144;
  float* vecs2 = scal2 + 12288;
  float* vn2   = vecs2 + 18432;
  float* pacc  = vn2 + 6144;             // 1024
  float* vacc  = pacc + 1024;            // 3072

  prep_kernel<<<(ZERO_FLOATS + 255) / 256, 256, 0, stream>>>(normals, wsf);

  const dim3 grid(16 * NB * NS);         // 16 target tiles x (b,s); 64 targets/block
  const dim3 blk(BLOCK);

  // layer 0 (scal/vecs/vn == 0)
  globe_pass<false><<<grid, blk, 0, stream>>>(
      centroids, centroids, nrmW, areas, refL,
      scal0, vecs0, vn0,
      cW1, cb1, cW2, cb2, cW3, cb3, cWo, cbo,
      scal1, vecs1);
  vn_kernel<<<(NTGT * NLV + 255) / 256, 256, 0, stream>>>(vecs1, nrmW, vn1);

  // layer 1 (weights offset by layer stride)
  const int sW1 = NB * NS * FF * HH, sB = NB * NS * HH;
  const int sW2 = NB * NS * HH * HH;
  const int sWo = NB * NS * HH * 24, sBo = NB * NS * 24;
  globe_pass<false><<<grid, blk, 0, stream>>>(
      centroids, centroids, nrmW, areas, refL,
      scal1, vecs1, vn1,
      cW1 + sW1, cb1 + sB, cW2 + sW2, cb2 + sB, cW3 + sW2, cb3 + sB, cWo + sWo, cbo + sBo,
      scal2, vecs2);
  vn_kernel<<<(NTGT * NLV + 255) / 256, 256, 0, stream>>>(vecs2, nrmW, vn2);

  // final prediction pass
  globe_pass<true><<<grid, blk, 0, stream>>>(
      pred, centroids, nrmW, areas, refL,
      scal2, vecs2, vn2,
      fW1, fb1, fW2, fb2, fW3, fb3, fWo, fbo,
      pacc, vacc);

  calib_kernel<<<(NPTS + 255) / 256, 256, 0, stream>>>(pacc, vacc, calib, (float*)d_out);
}
